// MHA_82025285419533
// MI455X (gfx1250) — matrix operand order repaired and checked by reading, not compiled, not run on hardware
//
#include <hip/hip_runtime.h>

// MHA forward for MI455X (gfx1250, wave32, WMMA 16x16x32 f16 + TDM staging).
// B=8, S=1024, E=1024, H=16, D=64, BH=128, M=B*S=8192.

typedef __attribute__((ext_vector_type(16))) _Float16 v16h;
typedef __attribute__((ext_vector_type(8)))  _Float16 v8h;
typedef __attribute__((ext_vector_type(8)))  float    v8f;
typedef unsigned int u32x4 __attribute__((ext_vector_type(4)));
typedef int          i32x4 __attribute__((ext_vector_type(4)));
typedef int          i32x8 __attribute__((ext_vector_type(8)));

#define S_LEN   1024
#define E_DIM   1024
#define HDIM    64
#define BH      128          // B*H
#define MROWS   8192         // B*S
#define ELEM8M  (8u*1024u*1024u)
#define KC      64           // K-chunk staged by TDM

// ---- fragment loaders -------------------------------------------------------
// A-matrix 16x32 f16: lanes 0-15 hold K 0..15,
// lanes 16-31 hold K 16..31 (row = lane%16).
__device__ __forceinline__ v16h load_a_frag(const _Float16* __restrict__ p, int half) {
    return *(const v16h*)(p + half * 16);
}
// B-matrix 32x16 f16: lanes 0-15 hold K 0..15, lanes 16-31 hold K 16..31.
__device__ __forceinline__ v16h load_b_frag(const _Float16* __restrict__ p, int half) {
    return *(const v16h*)(p + half * 16);
}

// ---- f32 -> f16 convert -----------------------------------------------------
__global__ void cvt_f32_to_f16(const float* __restrict__ src, _Float16* __restrict__ dst, int n) {
    int i = blockIdx.x * blockDim.x + threadIdx.x;
    if (i < n) dst[i] = (_Float16)src[i];
}

// ---- register-tiled batched NT GEMM (wave tile 64x32 = 4x2 WMMA tiles) ------
// blockDim = 128 (4 waves). grid.x*4 waves == (M/64)*(N/32); grid.y = batch.
__global__ void gemm_nt_f16_reg(const _Float16* __restrict__ A, int lda, long long strideA,
                                const _Float16* __restrict__ Bm, int ldb, long long strideB,
                                const float* __restrict__ bias,
                                float* __restrict__ Cf, _Float16* __restrict__ Ch,
                                int ldc, long long strideC,
                                int wtilesN, int K, float scale) {
    const int wave = threadIdx.x >> 5;
    const int lane = threadIdx.x & 31;
    const int r    = lane & 15;
    const int half = lane >> 4;
    const int wt   = blockIdx.x * 4 + wave;
    const int wm   = wt / wtilesN;            // 64-row stripe
    const int wn   = wt % wtilesN;            // 32-col stripe
    const long long batch = blockIdx.y;

    const _Float16* ap = A  + batch * strideA + (size_t)(wm * 64 + r) * lda;
    const _Float16* bp = Bm + batch * strideB + (size_t)(wn * 32 + r) * ldb;

    v8f acc[4][2] = {};
    for (int k0 = 0; k0 < K; k0 += 32) {
        v16h af[4], bf[2];
#pragma unroll
        for (int i = 0; i < 4; ++i) af[i] = load_a_frag(ap + (size_t)i * 16 * lda + k0, half);
#pragma unroll
        for (int j = 0; j < 2; ++j) bf[j] = load_b_frag(bp + (size_t)j * 16 * ldb + k0, half);
#pragma unroll
        for (int i = 0; i < 4; ++i)
#pragma unroll
            for (int j = 0; j < 2; ++j)
                acc[i][j] = __builtin_amdgcn_wmma_f32_16x16x32_f16(false, af[i], false, bf[j],
                                                                   (short)0, acc[i][j], false, false);
    }
#pragma unroll
    for (int j = 0; j < 2; ++j) {
        const int   n  = wn * 32 + j * 16 + r;
        const float bn = bias ? bias[n] : 0.0f;
#pragma unroll
        for (int i = 0; i < 4; ++i)
#pragma unroll
            for (int e = 0; e < 8; ++e) {
                const int m = wm * 64 + i * 16 + half * 8 + e;
                const float v = acc[i][j][e] * scale + bn;
                const long long idx = batch * strideC + (long long)m * ldc + n;
                if (Cf) Cf[idx] = v;
                if (Ch) Ch[idx] = (_Float16)v;
            }
    }
}

// ---- TDM-staged NT GEMM for the big weight-stationary GEMMs -----------------
// Block = 256 threads (8 waves) computing a 128x128 C tile; waves arranged
// 4(m) x 2(n), each wave a 32x64 tile (2x4 WMMA tiles). The 128xKC B slab is
// DMA'd into LDS by the Tensor Data Mover; A streams from global (L2-resident).
__global__ void __launch_bounds__(256)
gemm_nt_f16_tdm(const _Float16* __restrict__ A, int lda,
                const _Float16* __restrict__ Bm, int ldb, int nrowsB,
                const float* __restrict__ bias,
                float* __restrict__ Cf, _Float16* __restrict__ Ch,
                int ldc, int blocksN, int K, float scale) {
    __shared__ __align__(32) _Float16 btile[128 * KC];   // 16 KB

    const int wave = threadIdx.x >> 5;
    const int lane = threadIdx.x & 31;
    const int r    = lane & 15;
    const int half = lane >> 4;
    const int bm   = blockIdx.x / blocksN;    // 128-row block
    const int bn   = blockIdx.x % blocksN;    // 128-col block
    const int wm   = wave >> 1;               // 0..3 -> 32 rows each
    const int wn   = wave & 1;                // 0..1 -> 64 cols each

    const _Float16* ap = A + (size_t)(bm * 128 + wm * 32 + r) * lda;
    const unsigned long long gbase =
        (unsigned long long)(uintptr_t)(Bm + (size_t)(bn * 128) * ldb);
    const unsigned int ldsbase = (unsigned int)(uintptr_t)btile;  // low 32 bits = LDS offset

    v8f acc[2][4] = {};
    for (int k0 = 0; k0 < K; k0 += KC) {
        if (wave == 0) {
            const unsigned long long ga = gbase + (unsigned long long)k0 * 2ull;
            // D# group 0: count=1 | lds_addr | global_addr | type=2
            u32x4 g0 = { 1u,
                         ldsbase,
                         (unsigned int)ga,
                         (unsigned int)(((ga >> 32) & 0x1FFFFFFull) | 0x80000000ull) };
            // D# group 1: data_size=1(2B); tensor_dim0=ldb; tensor_dim1=nrowsB;
            // tile_dim0=KC; tile_dim1=128; tensor_dim0_stride=ldb
            i32x8 g1 = { (int)0x00010000u,
                         (int)(((unsigned)ldb & 0xFFFFu) << 16),
                         (int)(((unsigned)ldb >> 16) | (((unsigned)nrowsB & 0xFFFFu) << 16)),
                         (int)(((unsigned)nrowsB >> 16) | ((unsigned)KC << 16)),
                         128,
                         ldb,
                         0, 0 };
            i32x4 gz = { 0, 0, 0, 0 };
#if defined(__clang_major__) && (__clang_major__ >= 23)
            i32x8 gz8 = { 0, 0, 0, 0, 0, 0, 0, 0 };
            __builtin_amdgcn_tensor_load_to_lds(g0, g1, gz, gz, gz8, 0);
#else
            __builtin_amdgcn_tensor_load_to_lds(g0, g1, gz, gz, 0);
#endif
            __builtin_amdgcn_s_wait_tensorcnt(0);
        }
        __syncthreads();
#pragma unroll
        for (int kk = 0; kk < KC; kk += 32) {
            v16h af[2], bf[4];
#pragma unroll
            for (int i = 0; i < 2; ++i)
                af[i] = load_a_frag(ap + (size_t)i * 16 * lda + k0 + kk, half);
#pragma unroll
            for (int j = 0; j < 4; ++j)
                bf[j] = *(const v16h*)(&btile[(wn * 64 + j * 16 + r) * KC + kk + half * 16]);
#pragma unroll
            for (int i = 0; i < 2; ++i)
#pragma unroll
                for (int j = 0; j < 4; ++j)
                    acc[i][j] = __builtin_amdgcn_wmma_f32_16x16x32_f16(false, af[i], false, bf[j],
                                                                       (short)0, acc[i][j], false, false);
        }
        __syncthreads();
    }
#pragma unroll
    for (int j = 0; j < 4; ++j) {
        const int   n  = bn * 128 + wn * 64 + j * 16 + r;
        const float bn_ = bias ? bias[n] : 0.0f;
#pragma unroll
        for (int i = 0; i < 2; ++i)
#pragma unroll
            for (int e = 0; e < 8; ++e) {
                const int m = bm * 128 + wm * 32 + i * 16 + half * 8 + e;
                const float v = acc[i][j][e] * scale + bn_;
                const long long idx = (long long)m * ldc + n;
                if (Cf) Cf[idx] = v;
                if (Ch) Ch[idx] = (_Float16)v;
            }
    }
}

// ---- softmax over rows of attn (in place), 256 threads/row, 1024 cols -------
__global__ void softmax_rows(float* __restrict__ attn) {
    __shared__ float red[256];
    float* row = attn + (size_t)blockIdx.x * S_LEN;
    const int t = threadIdx.x;
    float4 v = ((const float4*)row)[t];

    float m = fmaxf(fmaxf(v.x, v.y), fmaxf(v.z, v.w));
    red[t] = m; __syncthreads();
    for (int s = 128; s > 0; s >>= 1) { if (t < s) red[t] = fmaxf(red[t], red[t + s]); __syncthreads(); }
    m = red[0]; __syncthreads();

    v.x = __expf(v.x - m); v.y = __expf(v.y - m);
    v.z = __expf(v.z - m); v.w = __expf(v.w - m);
    red[t] = v.x + v.y + v.z + v.w; __syncthreads();
    for (int s = 128; s > 0; s >>= 1) { if (t < s) red[t] += red[t + s]; __syncthreads(); }
    const float inv = 1.0f / red[0];

    v.x *= inv; v.y *= inv; v.z *= inv; v.w *= inv;
    ((float4*)row)[t] = v;
}

// ---- V transpose per head: vt[bh, d, t] = v[t*8192 + bh*64 + d] -------------
__global__ void transpose_v(const _Float16* __restrict__ v, _Float16* __restrict__ vt) {
    const int o  = blockIdx.x * blockDim.x + threadIdx.x;   // 8M outputs
    const int t  = o & (S_LEN - 1);
    const int d  = (o >> 10) & (HDIM - 1);
    const int bh = o >> 16;
    vt[o] = v[(size_t)t * MROWS + bh * HDIM + d];
}

// ---- PV GEMM: ctx[bh,t,d] = sum_tk P[bh,t,tk] * Vt[bh,d,tk] -----------------
// Register tiled 4x2 like gemm_nt_f16_reg; A operand (P) is f32 -> f16 on fly.
__global__ void gemm_pv_reg(const float* __restrict__ P, int ldp, long long strideP,
                            const _Float16* __restrict__ Vt, int ldv, long long strideV,
                            _Float16* __restrict__ Ch, int ldc, long long strideC,
                            int wtilesN, int K) {
    const int wave = threadIdx.x >> 5;
    const int lane = threadIdx.x & 31;
    const int r    = lane & 15;
    const int half = lane >> 4;
    const int wt   = blockIdx.x * 4 + wave;
    const int wm   = wt / wtilesN;
    const int wn   = wt % wtilesN;
    const long long batch = blockIdx.y;

    const float*    pp = P  + batch * strideP + (size_t)(wm * 64 + r) * ldp;
    const _Float16* bp = Vt + batch * strideV + (size_t)(wn * 32 + r) * ldv;

    v8f acc[4][2] = {};
    for (int k0 = 0; k0 < K; k0 += 32) {
        v16h af[4], bf[2];
#pragma unroll
        for (int i = 0; i < 4; ++i) {
            const float* pr = pp + (size_t)i * 16 * ldp + k0;
            v8f lo = *(const v8f*)(pr + half * 16);
            v8f hi = *(const v8f*)(pr + half * 16 + 8);
#pragma unroll
            for (int j = 0; j < 8; ++j) af[i][j]     = (_Float16)lo[j];
#pragma unroll
            for (int j = 0; j < 8; ++j) af[i][8 + j] = (_Float16)hi[j];
        }
#pragma unroll
        for (int j = 0; j < 2; ++j) bf[j] = load_b_frag(bp + (size_t)j * 16 * ldv + k0, half);
#pragma unroll
        for (int i = 0; i < 4; ++i)
#pragma unroll
            for (int j = 0; j < 2; ++j)
                acc[i][j] = __builtin_amdgcn_wmma_f32_16x16x32_f16(false, af[i], false, bf[j],
                                                                   (short)0, acc[i][j], false, false);
    }
#pragma unroll
    for (int j = 0; j < 2; ++j) {
        const int n = wn * 32 + j * 16 + r;
#pragma unroll
        for (int i = 0; i < 4; ++i)
#pragma unroll
            for (int e = 0; e < 8; ++e) {
                const int m = wm * 64 + i * 16 + half * 8 + e;
                const long long idx = batch * strideC + (long long)m * ldc + n;
                Ch[idx] = (_Float16)acc[i][j][e];
            }
    }
}

extern "C" void kernel_launch(void* const* d_in, const int* in_sizes, int n_in,
                              void* d_out, int out_size, void* d_ws, size_t ws_size,
                              hipStream_t stream) {
    (void)in_sizes; (void)n_in; (void)out_size; (void)ws_size;

    const float* q_in  = (const float*)d_in[0];
    const float* k_in  = (const float*)d_in[1];
    const float* v_in  = (const float*)d_in[2];
    const float* w_in  = (const float*)d_in[3];   // (3E, E)
    const float* b_in  = (const float*)d_in[4];   // (3E,)
    const float* w_out = (const float*)d_in[5];   // (E, E)
    const float* b_out = (const float*)d_in[6];   // (E,)

    float* out  = (float*)d_out;                  // (S, B, E)  = 8M f32
    float* attn = out + (size_t)ELEM8M;           // (BH, S, S) = 128M f32

    _Float16* w16    = (_Float16*)d_ws;
    _Float16* q16i   = w16;                       // 8M
    _Float16* k16i   = q16i  + ELEM8M;            // 8M
    _Float16* v16i   = k16i  + ELEM8M;            // 8M
    _Float16* win16  = v16i  + ELEM8M;            // 3M
    _Float16* wout16 = win16 + 3u*1024u*1024u;    // 1M
    _Float16* q16    = wout16+ 1024u*1024u;       // 8M
    _Float16* k16    = q16   + ELEM8M;            // 8M
    _Float16* v16    = k16   + ELEM8M;            // 8M
    _Float16* vt16   = v16   + ELEM8M;            // 8M
    _Float16* ctx16  = vt16  + ELEM8M;            // 8M

    // 1) convert inputs/weights to f16
    cvt_f32_to_f16<<<ELEM8M/256, 256, 0, stream>>>(q_in, q16i, ELEM8M);
    cvt_f32_to_f16<<<ELEM8M/256, 256, 0, stream>>>(k_in, k16i, ELEM8M);
    cvt_f32_to_f16<<<ELEM8M/256, 256, 0, stream>>>(v_in, v16i, ELEM8M);
    cvt_f32_to_f16<<<(3u*1024u*1024u)/256, 256, 0, stream>>>(w_in, win16, 3*1024*1024);
    cvt_f32_to_f16<<<(1024u*1024u)/256, 256, 0, stream>>>(w_out, wout16, 1024*1024);

    // 2) in-projections via TDM-staged GEMM: (8192x1024) @ (1024x1024)^T
    {
        dim3 grid((MROWS/128) * (E_DIM/128));   // 64*8 = 512 blocks
        gemm_nt_f16_tdm<<<grid, 256, 0, stream>>>(q16i, E_DIM, win16,                 E_DIM, E_DIM,
                                                  b_in,        nullptr, q16, E_DIM,
                                                  E_DIM/128, E_DIM, 1.0f);
        gemm_nt_f16_tdm<<<grid, 256, 0, stream>>>(k16i, E_DIM, win16 + 1024u*1024u,   E_DIM, E_DIM,
                                                  b_in + 1024, nullptr, k16, E_DIM,
                                                  E_DIM/128, E_DIM, 1.0f);
        gemm_nt_f16_tdm<<<grid, 256, 0, stream>>>(v16i, E_DIM, win16 + 2u*1024u*1024u, E_DIM, E_DIM,
                                                  b_in + 2048, nullptr, v16, E_DIM,
                                                  E_DIM/128, E_DIM, 1.0f);
    }

    // 3) per-head scores: attn[bh] = (Q_h @ K_h^T) / 8, f32 into d_out
    {
        dim3 grid((S_LEN/64) * (S_LEN/32) / 4, BH);   // 128 x 128
        gemm_nt_f16_reg<<<grid, 128, 0, stream>>>(q16, MROWS, HDIM, k16, MROWS, HDIM,
                                                  nullptr, attn, nullptr,
                                                  S_LEN, (long long)S_LEN * S_LEN,
                                                  S_LEN/32, HDIM, 0.125f);
    }

    // 4) softmax in place over attn rows
    softmax_rows<<<BH * S_LEN, 256, 0, stream>>>(attn);

    // 5) transpose V per head for K-contiguous B operand
    transpose_v<<<ELEM8M/256, 256, 0, stream>>>(v16, vt16);

    // 6) ctx = P @ V per head (f16 out, scattered to flat (T, BH, D))
    {
        dim3 grid((S_LEN/64) * (HDIM/32) / 4, BH);    // 8 x 128
        gemm_pv_reg<<<grid, 128, 0, stream>>>(attn, S_LEN, (long long)S_LEN * S_LEN,
                                              vt16, S_LEN, (long long)HDIM * S_LEN,
                                              ctx16, MROWS, HDIM,
                                              HDIM/32, S_LEN);
    }

    // 7) out-projection via TDM-staged GEMM: out = ctx @ w_out^T + b_out
    {
        dim3 grid((MROWS/128) * (E_DIM/128));
        gemm_nt_f16_tdm<<<grid, 256, 0, stream>>>(ctx16, E_DIM, wout16, E_DIM, E_DIM,
                                                  b_out, out, nullptr, E_DIM,
                                                  E_DIM/128, E_DIM, 1.0f);
    }
}
